// Quantizer_80255758893332
// MI455X (gfx1250) — matrix operand order repaired and checked by reading, not compiled, not run on hardware
//
#include <hip/hip_runtime.h>
#include <hip/hip_bf16.h>

typedef __attribute__((ext_vector_type(16))) __bf16 v16bf;
typedef __attribute__((ext_vector_type(8)))  float  v8f;

#define DIM      64
#define KCODES   1024
#define NTILES   (KCODES / 16)      // 64 code tiles of 16
#define NROWS    (64 * 4096)        // 262144
#define RT       4                  // M-tiles (of 16 rows) per wave
#define ROWS_PER_WAVE   (RT * 16)   // 64
#define WAVES_PER_BLOCK 4
#define ROWS_PER_BLOCK  (ROWS_PER_WAVE * WAVES_PER_BLOCK)  // 256
#define SBIAS    512.0f             // keeps biased scores strictly positive

union AFrag { v16bf v; __bf16 e[16]; };
union BFrag { v16bf v; uint4  q[2];  };

__device__ __forceinline__ void cvt8(const float* __restrict__ p, __bf16* d) {
    float4 f0 = *(const float4*)p;
    float4 f1 = *(const float4*)(p + 4);
    d[0] = (__bf16)f0.x; d[1] = (__bf16)f0.y; d[2] = (__bf16)f0.z; d[3] = (__bf16)f0.w;
    d[4] = (__bf16)f1.x; d[5] = (__bf16)f1.y; d[6] = (__bf16)f1.z; d[7] = (__bf16)f1.w;
}

// One block (32 lanes) per code: emit bf16 row + biased squared L2 norm.
__global__ void vq_prep_kernel(const float* __restrict__ emb,
                               __bf16* __restrict__ ebf,
                               float* __restrict__ nbias) {
    const int code = blockIdx.x;
    const int lane = threadIdx.x;
    const float* row = emb + code * DIM;
    float s = 0.0f;
#pragma unroll
    for (int i = 0; i < 2; ++i) {
        float v = row[lane + i * 32];
        ebf[code * DIM + lane + i * 32] = (__bf16)v;
        s = __builtin_fmaf(v, v, s);
    }
#pragma unroll
    for (int off = 16; off > 0; off >>= 1) s += __shfl_xor(s, off, 32);
    if (lane == 0) nbias[code] = s + SBIAS;
}

// Issue the B-fragment + biased-norm loads for code tile t.
// B layout (32x16): lanes 0-15 -> col=lane,   K = {c*32+0..7, c*32+16..23};
//                   lanes 16-31 -> col=lane-16, same + 8.
__device__ __forceinline__ void load_tile(const __bf16* __restrict__ ebf,
                                          const float* __restrict__ nbias,
                                          int t, int l15, int half,
                                          BFrag& b0, BFrag& b1, float& nb) {
    const int code = t * 16 + l15;
    const __bf16* bp = ebf + (size_t)code * DIM + half * 8;
    b0.q[0] = *(const uint4*)(bp);
    b0.q[1] = *(const uint4*)(bp + 16);
    b1.q[0] = *(const uint4*)(bp + 32);
    b1.q[1] = *(const uint4*)(bp + 48);
    nb = nbias[code];
}

__global__ __launch_bounds__(WAVES_PER_BLOCK * 32)
void vq_main_kernel(const float*  __restrict__ z,
                    const float*  __restrict__ emb,
                    const __bf16* __restrict__ ebf,
                    const float*  __restrict__ nbias,
                    float* __restrict__ outq,
                    int*   __restrict__ outidx) {
    const int lane    = threadIdx.x & 31;
    const int wave    = threadIdx.x >> 5;
    const int rowbase = blockIdx.x * ROWS_PER_BLOCK + wave * ROWS_PER_WAVE;
    const int half    = lane >> 4;      // 0: lanes 0-15, 1: lanes 16-31
    const int l15     = lane & 15;

    // ---- Build A fragments (16x32 bf16) for RT row tiles x 2 K-chunks ----
    // ISA A layout: lanes 0-15 hold row M=lane, K = {c*32+0..7, c*32+16..23};
    //               lanes 16-31 hold row M=lane-16, same + 8.
    AFrag a[RT][2];
#pragma unroll
    for (int rt = 0; rt < RT; ++rt) {
        const float* zr = z + (size_t)(rowbase + rt * 16 + l15) * DIM;
#pragma unroll
        for (int c = 0; c < 2; ++c) {
            const float* p0 = zr + c * 32 + half * 8;
            cvt8(p0,      a[rt][c].e);
            cvt8(p0 + 16, a[rt][c].e + 8);
        }
    }

    // ---- Running argmin as packed u32 keys: [score-bits & ~1023 | code] ----
    // Biased score is strictly positive -> float order == unsigned bit order.
    unsigned int bk[RT][8];
#pragma unroll
    for (int rt = 0; rt < RT; ++rt)
#pragma unroll
        for (int r = 0; r < 8; ++r) bk[rt][r] = 0xFFFFFFFFu;

    // ---- Double-buffered sweep of the 64 code tiles (2 per iteration) ----
    BFrag b0[2], b1[2];
    float nb[2];
    load_tile(ebf, nbias, 0, l15, half, b0[0], b1[0], nb[0]);

    for (int t = 0; t < NTILES; t += 2) {
#pragma unroll
        for (int u = 0; u < 2; ++u) {
            // Prefetch the next tile into the other buffer while computing.
            int tn = t + u + 1;
            if (tn > NTILES - 1) tn = NTILES - 1;   // harmless redundant load
            load_tile(ebf, nbias, tn, l15, half,
                      b0[u ^ 1], b1[u ^ 1], nb[u ^ 1]);

            const unsigned int code = (t + u) * 16 + l15;

            // Phase 1: K-chunk 0 for all row tiles (4 independent chains).
            v8f acc[RT];
#pragma unroll
            for (int rt = 0; rt < RT; ++rt)
                acc[rt] = __builtin_amdgcn_wmma_f32_16x16x32_bf16(
                    false, a[rt][0].v, false, b0[u].v, (short)0, v8f{}, false, false);

            // Phase 2: K-chunk 1 accumulate (each RAW is 3 WMMAs away).
#pragma unroll
            for (int rt = 0; rt < RT; ++rt)
                acc[rt] = __builtin_amdgcn_wmma_f32_16x16x32_bf16(
                    false, a[rt][1].v, false, b1[u].v, (short)0, acc[rt], false, false);

            // Phase 3: packed-key min updates (fills WMMA->VALU hazard slots).
#pragma unroll
            for (int rt = 0; rt < RT; ++rt) {
#pragma unroll
                for (int r = 0; r < 8; ++r) {
                    float s = __builtin_fmaf(-2.0f, acc[rt][r], nb[u]); // > 0
                    unsigned int key = (__float_as_uint(s) & 0xFFFFFC00u) | code;
                    bk[rt][r] = (key < bk[rt][r]) ? key : bk[rt][r];    // v_min_u32
                }
            }
        }
    }

    // ---- Cross-lane min within each 16-lane half (slot r holds row M=r for
    //      lanes 0-15, M=r+8 for lanes 16-31); low bits break ties by index ----
#pragma unroll
    for (int rt = 0; rt < RT; ++rt)
#pragma unroll
        for (int r = 0; r < 8; ++r) {
            unsigned int k = bk[rt][r];
#pragma unroll
            for (int off = 8; off > 0; off >>= 1) {
                unsigned int pk = __shfl_xor(k, off, 16);
                k = (pk < k) ? pk : k;
            }
            bk[rt][r] = k;
        }

    // ---- Write indices (lane 0 owns rows 0-7 of each tile, lane 16 rows 8-15) ----
    if (l15 == 0) {
#pragma unroll
        for (int rt = 0; rt < RT; ++rt)
#pragma unroll
            for (int r = 0; r < 8; ++r)
                outidx[rowbase + rt * 16 + half * 8 + r] = (int)(bk[rt][r] & 1023u);
    }

    // ---- Gather z_q rows from the exact f32 codebook (float2 per lane) ----
#pragma unroll
    for (int rt = 0; rt < RT; ++rt) {
#pragma unroll
        for (int m = 0; m < 16; ++m) {
            unsigned int idx =
                __shfl(bk[rt][m & 7], (m < 8) ? 0 : 16, 32) & 1023u;
            const float2* src = (const float2*)(emb + (size_t)idx * DIM);
            float2* dst = (float2*)(outq + (size_t)(rowbase + rt * 16 + m) * DIM);
            dst[lane] = src[lane];
        }
    }
}

extern "C" void kernel_launch(void* const* d_in, const int* in_sizes, int n_in,
                              void* d_out, int out_size, void* d_ws, size_t ws_size,
                              hipStream_t stream) {
    const float* z   = (const float*)d_in[0];   // [64,4096,64] f32
    const float* emb = (const float*)d_in[1];   // [1024,64]   f32

    __bf16* ebf   = (__bf16*)d_ws;                                        // 128 KB
    float*  nbias = (float*)((char*)d_ws + KCODES * DIM * sizeof(__bf16)); // 4 KB

    float* outq   = (float*)d_out;                       // first: N*D floats
    int*   outidx = (int*)d_out + (size_t)NROWS * DIM;   // then: N int32 indices

    vq_prep_kernel<<<KCODES, 32, 0, stream>>>(emb, ebf, nbias);
    vq_main_kernel<<<NROWS / ROWS_PER_BLOCK, WAVES_PER_BLOCK * 32, 0, stream>>>(
        z, emb, ebf, nbias, outq, outidx);
}
